// SemLA_2422361554994
// MI455X (gfx1250) — matrix operand order repaired and checked by reading, not compiled, not run on hardware
//
#include <hip/hip_runtime.h>
#include <hip/hip_bf16.h>

#define HW   1200   // 30*40 spatial positions
#define CDIM 1024   // channels
#define WDIM 40
#define BT   48     // 48x48 register-blocked macro-tile per wave (1200 = 48*25)
#define NB   25     // blocks per GEMM dimension

typedef __attribute__((ext_vector_type(16))) __bf16 bf16x16;
typedef __attribute__((ext_vector_type(8)))  __bf16 bf16x8;
typedef __attribute__((ext_vector_type(8)))  float  f32x8;

// --- 1) transpose [C, HW] -> [HW, C] row-major bf16, folding 1/sqrt(C) ---
__global__ __launch_bounds__(256)
void semla_cvt(const float* __restrict__ vi, const float* __restrict__ ir,
               __bf16* __restrict__ vbf, __bf16* __restrict__ ibf) {
  const float scale = 0.03125f;  // 1/sqrt(1024)
  int idx = blockIdx.x * 256 + threadIdx.x;
  if (idx >= HW * CDIM) return;
  int p = idx / CDIM;
  int c = idx - p * CDIM;
  vbf[idx] = (__bf16)(vi[c * HW + p] * scale);
  ibf[idx] = (__bf16)(ir[c * HW + p] * scale);
}

// A 16x32 bf16 fragment loader (ISA 7.12.2 16-bit A layout):
// lanes 0-15 hold K {k..k+7, k+16..k+23} of row M=r,
// lanes 16-31 hold K {k+8..k+15, k+24..k+31}  -> two b128 chunks per lane.
__device__ __forceinline__ bf16x16 load_afrag(const __bf16* p) {
  bf16x8 lo = *(const bf16x8*)p;
  bf16x8 hi = *(const bf16x8*)(p + 16);
  return __builtin_shufflevector(lo, hi, 0, 1, 2, 3, 4, 5, 6, 7,
                                         8, 9, 10, 11, 12, 13, 14, 15);
}

// --- 2) conf = (fv . fi^T) / TEMP, 48x48 tile per wave32 -----------------
// 3 A frags + 3 B frags -> 9 v_wmma_f32_16x16x32_bf16 per k-step.
// L2 intensity: 9*16384 FLOP per 6 KB operands = 24 FLOP/B (3x the naive tile).
__global__ __launch_bounds__(32)
void semla_gemm(const __bf16* __restrict__ A,   // [HW][CDIM] fv (bf16)
                const __bf16* __restrict__ B,   // [HW][CDIM] fi (bf16)
                float* __restrict__ conf) {     // [HW][HW]
  const int lane = threadIdx.x;        // 0..31
  const int r    = lane & 15;
  const int hi   = lane >> 4;          // lane group 0/1
  const int tm   = blockIdx.x * BT;    // row (l) base
  const int tn   = blockIdx.y * BT;    // col (s) base

  const __bf16* ap = A + (size_t)(tm + r) * CDIM + hi * 8;
  // B 32x16 fragment: lane n = r is column N; lanes 0-15 K={k..k+7,k+16..k+23},
  // lanes 16-31 K={k+8..k+15,k+24..k+31} (same order as A).
  const __bf16* bp = B + (size_t)(tn + r) * CDIM + hi * 8;

  f32x8 acc[3][3] = {};
#pragma unroll 2
  for (int k = 0; k < CDIM; k += 32) {
    bf16x16 a[3], b[3];
#pragma unroll
    for (int i = 0; i < 3; ++i) {
      a[i] = load_afrag(ap + (size_t)(16 * i) * CDIM + k);
      b[i] = load_afrag(bp + (size_t)(16 * i) * CDIM + k);
    }
#pragma unroll
    for (int i = 0; i < 3; ++i)
#pragma unroll
      for (int j = 0; j < 3; ++j)
        acc[i][j] = __builtin_amdgcn_wmma_f32_16x16x32_bf16(
            /*neg_a=*/false, a[i], /*neg_b=*/false, b[j],
            /*c_mod=*/(short)0, acc[i][j],
            /*reuse_a=*/false, /*reuse_b=*/false);
  }

  // C/D layout: lane L -> N = L%16, M = e + 8*(L/16) for acc element e
  const float inv_temp = 10.0f;  // 1 / TEMP
#pragma unroll
  for (int i = 0; i < 3; ++i) {
#pragma unroll
    for (int j = 0; j < 3; ++j) {
      float* out = conf + (size_t)(tm + 16 * i + hi * 8) * HW + (tn + 16 * j + r);
#pragma unroll
      for (int e = 0; e < 8; ++e) out[(size_t)e * HW] = acc[i][j][e] * inv_temp;
    }
  }
}

// --- 3) row-wise and column-wise maxima ----------------------------------
__global__ __launch_bounds__(256)
void semla_maxred(const float* __restrict__ conf,
                  float* __restrict__ rowmax, float* __restrict__ colmax) {
  int t = blockIdx.x * 256 + threadIdx.x;
  if (t < HW) {                       // max over s (axis 2)
    const float* row = conf + (size_t)t * HW;
    float m = row[0];
    for (int s = 1; s < HW; ++s) m = fmaxf(m, row[s]);
    rowmax[t] = m;
  } else if (t < 2 * HW) {            // max over l (axis 1)
    int s = t - HW;
    float m = conf[s];
    for (int l = 1; l < HW; ++l) m = fmaxf(m, conf[(size_t)l * HW + s]);
    colmax[s] = m;
  }
}

// --- 4) mutual-NN match + keypoint emission (int32) ----------------------
__global__ __launch_bounds__(256)
void semla_match(const float* __restrict__ conf,
                 const float* __restrict__ rowmax,
                 const float* __restrict__ colmax,
                 int* __restrict__ out) {        // [2400] mkpts0 | [2400] mkpts1
  int l = blockIdx.x * 256 + threadIdx.x;
  if (l >= HW) return;
  const float rm = rowmax[l];
  const float* row = conf + (size_t)l * HW;
  int j = 0, valid = 0;
  for (int s = 0; s < HW; ++s) {
    float v = row[s];
    if (v == rm && v == colmax[s]) { j = s; valid = 1; break; }  // first hit
  }
  int x = l % WDIM, y = l / WDIM;
  int jx = j % WDIM, jy = j / WDIM;
  out[2 * l + 0]          = valid ? x * 8 : 0;   // mkpts0_final
  out[2 * l + 1]          = valid ? y * 8 : 0;
  out[2 * HW + 2 * l + 0] = valid ? jx * 8 : 0;  // mkpts1_final
  out[2 * HW + 2 * l + 1] = valid ? jy * 8 : 0;
}

extern "C" void kernel_launch(void* const* d_in, const int* in_sizes, int n_in,
                              void* d_out, int out_size, void* d_ws, size_t ws_size,
                              hipStream_t stream) {
  const float* vi = (const float*)d_in[0];   // feat_vi [1,1024,30,40] f32
  const float* ir = (const float*)d_in[1];   // feat_ir [1,1024,30,40] f32
  int* out = (int*)d_out;                    // int32 outputs, concatenated

  char* ws = (char*)d_ws;
  __bf16* vbf   = (__bf16*)ws;                                    // 2.46 MB
  __bf16* ibf   = (__bf16*)(ws + (size_t)HW * CDIM * 2);          // 2.46 MB
  float*  conf  = (float*)(ws + (size_t)2 * HW * CDIM * 2);       // 5.76 MB
  float*  rowmax = conf + (size_t)HW * HW;
  float*  colmax = rowmax + HW;

  semla_cvt<<<(HW * CDIM + 255) / 256, 256, 0, stream>>>(vi, ir, vbf, ibf);

  dim3 grid(NB, NB);
  semla_gemm<<<grid, 32, 0, stream>>>(vbf, ibf, conf);

  semla_maxred<<<(2 * HW + 255) / 256, 256, 0, stream>>>(conf, rowmax, colmax);

  semla_match<<<(HW + 255) / 256, 256, 0, stream>>>(conf, rowmax, colmax, out);
}
